// MultiHeadAttention_48421461295288
// MI455X (gfx1250) — matrix operand order repaired and checked by reading, not compiled, not run on hardware
//
#include <hip/hip_runtime.h>
#include <hip/hip_bf16.h>
#include <stdint.h>

// ---------------------------------------------------------------------------
// MHA for MI455X (gfx1250, wave32). bf16 WMMA (v_wmma_f32_16x16x32_bf16),
// fp32 accumulate, fp32 softmax, TDM (tensor_load_to_lds) K/V tile staging.
// ---------------------------------------------------------------------------

typedef __attribute__((ext_vector_type(16))) __bf16 v16bf;
typedef __attribute__((ext_vector_type(8)))  __bf16 v8bf;
typedef __attribute__((ext_vector_type(8)))  float  v8f;

#define BQ 2
#define LQ 2048
#define DM 1024
#define NH 16
#define HD 64

#define KPAD 72   // K tile LDS row stride (bf16): 32 dwords + 4 pad dwords
#define VPAD 40   // V tile LDS row stride (bf16): 16 dwords + 4 pad dwords

#if defined(__AMDGCN__) && __has_builtin(__builtin_amdgcn_tensor_load_to_lds) && \
    __has_builtin(__builtin_amdgcn_s_wait_tensorcnt)
#define HAVE_TDM 1
#else
#define HAVE_TDM 0
#endif

__device__ __forceinline__ __bf16 f2bf(float f) {
    unsigned u = __builtin_bit_cast(unsigned, f);
    unsigned r = u + 0x7FFFu + ((u >> 16) & 1u);   // round-to-nearest-even
    unsigned short h = (unsigned short)(r >> 16);
    return __builtin_bit_cast(__bf16, h);
}

// --------------------------- fp32 -> bf16 convert ---------------------------
__global__ void cvt_f32_bf16(const float* __restrict__ x, __bf16* __restrict__ y, int n) {
    int i = blockIdx.x * blockDim.x + threadIdx.x;
    int stride = gridDim.x * blockDim.x;
    for (; i < n; i += stride) y[i] = f2bf(x[i]);
}

// ----------------------- WMMA fragment loaders (NT) -------------------------
// A fragment (16x32 bf16): lane l<16 -> row M=l, K in {kb..kb+7, kb+16..kb+23},
// kb=0; lanes 16-31 -> same rows, kb=8. Two contiguous 16B loads per lane.
__device__ __forceinline__ v16bf load_frag_a(const __bf16* base, long rowStride,
                                             int row0, int k0, int lane) {
    int m  = lane & 15;
    int kb = k0 + ((lane >> 4) << 3);
    const __bf16* p = base + (long)(row0 + m) * rowStride + kb;
    v8bf lo = *(const v8bf*)p;
    v8bf hi = *(const v8bf*)(p + 16);
    return __builtin_shufflevector(lo, hi, 0,1,2,3,4,5,6,7,8,9,10,11,12,13,14,15);
}

// B fragment (32x16 bf16 from row-major [N,K]): lane n<16 -> col N=n,
// K in {kb..kb+7, kb+16..kb+23}, kb=k0; lanes 16-31 -> kb=k0+8.
__device__ __forceinline__ v16bf load_frag_b(const __bf16* base, long rowStride,
                                             int col0, int k0, int lane) {
    int nn = lane & 15;
    int kb = k0 + ((lane >> 4) << 3);
    const __bf16* p = base + (long)(col0 + nn) * rowStride + kb;
    v8bf lo = *(const v8bf*)p;
    v8bf hi = *(const v8bf*)(p + 16);
    return __builtin_shufflevector(lo, hi, 0,1,2,3,4,5,6,7,8,9,10,11,12,13,14,15);
}

// --------------------------- TDM tile loader --------------------------------
#if HAVE_TDM
typedef unsigned int u32x4 __attribute__((ext_vector_type(4)));
typedef int          i32x4 __attribute__((ext_vector_type(4)));
typedef int          i32x8 __attribute__((ext_vector_type(8)));

// 2D bf16 tile load: tile_d0 elements per row, tile_d1 rows, row stride
// `stride0` elements; LDS row padding via D# pad_interval/pad_amount codes.
__device__ __forceinline__ void tdm_load_2d_bf16(unsigned lds_byte_off, const void* gtile,
                                                 unsigned tile_d0, unsigned tile_d1,
                                                 unsigned long long stride0,
                                                 unsigned pad_interval_code,
                                                 unsigned pad_amount_code)
{
    unsigned long long ga = (unsigned long long)(uintptr_t)gtile;
    u32x4 g0;
    g0.x = 1u;                                   // count=1, user descriptor
    g0.y = lds_byte_off;                         // lds_addr (bytes)
    g0.z = (unsigned)ga;                         // global_addr[31:0]
    g0.w = (unsigned)(ga >> 32) | (2u << 30);    // global_addr[56:32] | type=2
    i32x8 g1;
    g1[0] = (int)((1u << 16)                     // data_size = 2 bytes
                | (1u << 20)                     // pad_enable
                | (pad_interval_code << 22)
                | (pad_amount_code  << 25));
    g1[1] = (int)((tile_d0 & 0xFFFFu) << 16);                                  // tensor_dim0[15:0]
    g1[2] = (int)(((tile_d0 >> 16) & 0xFFFFu) | ((tile_d1 & 0xFFFFu) << 16));  // td0 hi | td1 lo
    g1[3] = (int)(((tile_d1 >> 16) & 0xFFFFu) | ((tile_d0 & 0xFFFFu) << 16));  // td1 hi | tile_dim0
    g1[4] = (int)(tile_d1 & 0xFFFFu);                                          // tile_dim1, tile_dim2=0
    g1[5] = (int)(unsigned)(stride0 & 0xFFFFFFFFull);                          // dim0 stride lo
    g1[6] = (int)((unsigned)(stride0 >> 32) & 0xFFFFu);                        // dim0 stride hi
    g1[7] = 0;
    i32x4 gz = {0, 0, 0, 0};
#if __has_include(<hip/amd_detail/amd_gfx1250_TDM.h>)
    i32x8 z8 = {0, 0, 0, 0, 0, 0, 0, 0};
    __builtin_amdgcn_tensor_load_to_lds(g0, g1, gz, gz, z8, 0);
#else
    __builtin_amdgcn_tensor_load_to_lds(g0, g1, gz, gz, 0);
#endif
}
#endif

// ------------------------------- GEMM (NT) ----------------------------------
// out[r,c] = sum_k A[r,k] * Bw[c,k] + bias[c]
// mode 0: bf16 head-split [B,H,L,HD]; mode 1: bf16 [B,H,HD,L]; mode 2: fp32 [M,N]
__global__ void __launch_bounds__(256)
gemm_nt_bf16(const __bf16* __restrict__ A, const __bf16* __restrict__ Bw,
             const float* __restrict__ bias, __bf16* __restrict__ outb,
             float* __restrict__ outf, int M, int N, int K, int mode)
{
    const int lane = threadIdx.x & 31;
    const int w    = threadIdx.x >> 5;
    const int wm   = (w >> 2) * 64;
    const int wn   = (w & 3) * 32;
    const int bn   = blockIdx.x * 128;
    const int bm   = blockIdx.y * 128;

    v8f acc[4][2];
#pragma unroll
    for (int i = 0; i < 4; ++i)
#pragma unroll
        for (int j = 0; j < 2; ++j) acc[i][j] = (v8f)(0.0f);

    for (int k0 = 0; k0 < K; k0 += 32) {
        v16bf af[4], bfr[2];
#pragma unroll
        for (int i = 0; i < 4; ++i)
            af[i] = load_frag_a(A, K, bm + wm + i * 16, k0, lane);
#pragma unroll
        for (int j = 0; j < 2; ++j)
            bfr[j] = load_frag_b(Bw, K, bn + wn + j * 16, k0, lane);
#pragma unroll
        for (int i = 0; i < 4; ++i)
#pragma unroll
            for (int j = 0; j < 2; ++j)
                acc[i][j] = __builtin_amdgcn_wmma_f32_16x16x32_bf16(
                    false, af[i], false, bfr[j], (short)0, acc[i][j], false, false);
    }

    const int half8 = (lane >> 4) << 3;
    const int cl    = lane & 15;
#pragma unroll
    for (int i = 0; i < 4; ++i) {
#pragma unroll
        for (int j = 0; j < 2; ++j) {
            int   c  = bn + wn + j * 16 + cl;
            float bs = bias ? bias[c] : 0.0f;
#pragma unroll
            for (int e = 0; e < 8; ++e) {
                int   r   = bm + wm + i * 16 + half8 + e;
                float val = acc[i][j][e] + bs;
                if (mode == 2) {
                    outf[(long)r * N + c] = val;
                } else {
                    int b = r >> 11, l = r & (LQ - 1);
                    int h = c >> 6,  hd = c & (HD - 1);
                    long idx = (mode == 0)
                        ? ((long)(b * NH + h) * LQ + l) * HD + hd
                        : ((long)(b * NH + h) * HD + hd) * LQ + l;
                    outb[idx] = f2bf(val);
                }
            }
        }
    }
}

// --------------------------- flash attention --------------------------------
// grid: (L/128, B*H), block 128 (4 waves). Each wave owns 32 query rows.
// K/V tiles staged per-block in LDS (TDM double-buffered when available).
__global__ void __launch_bounds__(128)
flash_attn(const __bf16* __restrict__ qh, const __bf16* __restrict__ kh,
           const __bf16* __restrict__ vT, __bf16* __restrict__ attnout)
{
    __shared__ __attribute__((aligned(16))) __bf16 Pl[4][32][40];          // P re-layout
    __shared__ __attribute__((aligned(16))) __bf16 Kl[HAVE_TDM ? 2 : 1][32 * KPAD];
    __shared__ __attribute__((aligned(16))) __bf16 Vl[HAVE_TDM ? 2 : 1][HD * VPAD];

    const int lane = threadIdx.x & 31;
    const int w    = threadIdx.x >> 5;
    const int bh   = blockIdx.y;                 // b*NH + h
    const int q0   = blockIdx.x * 128 + w * 32;

    const __bf16* qb = qh + (long)bh * LQ * HD;
    const __bf16* kb = kh + (long)bh * LQ * HD;
    const __bf16* vb = vT + (long)bh * HD * LQ;

    const int half8 = (lane >> 4) << 3;
    const int cl    = lane & 15;

    v8f o[2][4];
#pragma unroll
    for (int i = 0; i < 2; ++i)
#pragma unroll
        for (int j = 0; j < 4; ++j) o[i][j] = (v8f)(0.0f);

    float mrun[2][8], lrun[2][8];
#pragma unroll
    for (int i = 0; i < 2; ++i)
#pragma unroll
        for (int e = 0; e < 8; ++e) { mrun[i][e] = -3.0e38f; lrun[i][e] = 0.0f; }

    // Q fragments live in registers for the whole key loop.
    v16bf qa[2][2];
#pragma unroll
    for (int fm = 0; fm < 2; ++fm)
#pragma unroll
        for (int ks = 0; ks < 2; ++ks)
            qa[fm][ks] = load_frag_a(qb, HD, q0 + fm * 16, ks * 32, lane);

    const float scale = 0.125f;   // 1/sqrt(64)

#if HAVE_TDM
    const unsigned klo[2] = { (unsigned)(uintptr_t)&Kl[0][0], (unsigned)(uintptr_t)&Kl[1][0] };
    const unsigned vlo[2] = { (unsigned)(uintptr_t)&Vl[0][0], (unsigned)(uintptr_t)&Vl[1][0] };
    if (w == 0) {   // prologue: chunk 0 into buffer 0
        tdm_load_2d_bf16(klo[0], kb, HD, 32, HD, 4, 3);      // 32 key rows x 64
        tdm_load_2d_bf16(vlo[0], vb, 32, HD, LQ, 3, 3);      // 64 hd rows x 32
    }
#endif

    for (int it = 0; it < LQ / 32; ++it) {
        const int kc = it * 32;
#if HAVE_TDM
        const int cur = it & 1;
        if (w == 0) __builtin_amdgcn_s_wait_tensorcnt(0);    // chunk `it` resident
        __syncthreads();                                     // publish to all waves
        if (w == 0 && it + 1 < LQ / 32) {                    // overlap DMA of it+1
            int nb = (it + 1) & 1;
            tdm_load_2d_bf16(klo[nb], kb + (long)(kc + 32) * HD, HD, 32, HD, 4, 3);
            tdm_load_2d_bf16(vlo[nb], vb + (kc + 32), 32, HD, LQ, 3, 3);
        }
#else
        const int cur = 0;
        __syncthreads();                                     // WAR vs prev readers
        {
            int t = threadIdx.x;
            for (int c = t; c < 256; c += 128) {             // K: 32 rows x 8 x 16B
                int r = c >> 3, j = c & 7;
                *(v8bf*)&Kl[0][r * KPAD + j * 8] =
                    *(const v8bf*)(kb + (long)(kc + r) * HD + j * 8);
            }
            for (int c = t; c < 256; c += 128) {             // V: 64 rows x 4 x 16B
                int r = c >> 2, j = c & 3;
                *(v8bf*)&Vl[0][r * VPAD + j * 8] =
                    *(const v8bf*)(vb + (long)r * LQ + kc + j * 8);
            }
        }
        __syncthreads();
#endif
        const __bf16* Kt = &Kl[cur][0];
        const __bf16* Vt = &Vl[cur][0];

        // ---- S = q @ k^T ----------------------------------------------
        v8f s[2][2];
#pragma unroll
        for (int fm = 0; fm < 2; ++fm)
#pragma unroll
            for (int fn = 0; fn < 2; ++fn) s[fm][fn] = (v8f)(0.0f);

        v16bf kf[2][2];
#pragma unroll
        for (int fn = 0; fn < 2; ++fn)
#pragma unroll
            for (int ks = 0; ks < 2; ++ks)
                kf[fn][ks] = load_frag_b(Kt, KPAD, fn * 16, ks * 32, lane);

#pragma unroll
        for (int fm = 0; fm < 2; ++fm)
#pragma unroll
            for (int fn = 0; fn < 2; ++fn)
#pragma unroll
                for (int ks = 0; ks < 2; ++ks)
                    s[fm][fn] = __builtin_amdgcn_wmma_f32_16x16x32_bf16(
                        false, qa[fm][ks], false, kf[fn][ks], (short)0,
                        s[fm][fn], false, false);

        // ---- online softmax update ------------------------------------
#pragma unroll
        for (int fm = 0; fm < 2; ++fm) {
#pragma unroll
            for (int e = 0; e < 8; ++e) {
                float a  = s[fm][0][e] * scale;
                float b2 = s[fm][1][e] * scale;
                float v  = fmaxf(a, b2);
                v = fmaxf(v, __shfl_xor(v, 1));
                v = fmaxf(v, __shfl_xor(v, 2));
                v = fmaxf(v, __shfl_xor(v, 4));
                v = fmaxf(v, __shfl_xor(v, 8));   // stays within 16-lane half
                float mnew = fmaxf(mrun[fm][e], v);
                float f    = __expf(mrun[fm][e] - mnew);
                mrun[fm][e] = mnew;
                float p0 = __expf(a - mnew);
                float p1 = __expf(b2 - mnew);
                s[fm][0][e] = p0;
                s[fm][1][e] = p1;
                float rs = p0 + p1;
                rs += __shfl_xor(rs, 1);
                rs += __shfl_xor(rs, 2);
                rs += __shfl_xor(rs, 4);
                rs += __shfl_xor(rs, 8);
                lrun[fm][e] = lrun[fm][e] * f + rs;
#pragma unroll
                for (int fh = 0; fh < 4; ++fh) o[fm][fh][e] *= f;
            }
        }

        // ---- stage P (C-layout -> A-layout) through LDS ---------------
#pragma unroll
        for (int fm = 0; fm < 2; ++fm)
#pragma unroll
            for (int fn = 0; fn < 2; ++fn)
#pragma unroll
                for (int e = 0; e < 8; ++e)
                    Pl[w][fm * 16 + half8 + e][fn * 16 + cl] = f2bf(s[fm][fn][e]);
        __syncthreads();

        v16bf pa[2];
#pragma unroll
        for (int fm = 0; fm < 2; ++fm) {
            const __bf16* pp = &Pl[w][fm * 16 + (lane & 15)][(lane >> 4) << 3];
            v8bf lo = *(const v8bf*)pp;
            v8bf hi = *(const v8bf*)(pp + 16);
            pa[fm] = __builtin_shufflevector(lo, hi,
                         0,1,2,3,4,5,6,7,8,9,10,11,12,13,14,15);
        }
        __syncthreads();

        // ---- O += P @ V -----------------------------------------------
#pragma unroll
        for (int fh = 0; fh < 4; ++fh) {
            v16bf vf = load_frag_b(Vt, VPAD, fh * 16, 0, lane);
#pragma unroll
            for (int fm = 0; fm < 2; ++fm)
                o[fm][fh] = __builtin_amdgcn_wmma_f32_16x16x32_bf16(
                    false, pa[fm], false, vf, (short)0, o[fm][fh], false, false);
        }
    }

    // ---- epilogue: normalize and store bf16 [B, L, D] -----------------
    const int b = bh >> 4;
    const int h = bh & 15;
#pragma unroll
    for (int fm = 0; fm < 2; ++fm) {
#pragma unroll
        for (int e = 0; e < 8; ++e) {
            float inv = 1.0f / lrun[fm][e];
            int   row = q0 + fm * 16 + half8 + e;
#pragma unroll
            for (int fh = 0; fh < 4; ++fh) {
                int  col = h * HD + fh * 16 + cl;
                long idx = ((long)b * LQ + row) * DM + col;
                attnout[idx] = f2bf(o[fm][fh][e] * inv);
            }
        }
    }
}

// ------------------------------- launcher -----------------------------------
extern "C" void kernel_launch(void* const* d_in, const int* in_sizes, int n_in,
                              void* d_out, int out_size, void* d_ws, size_t ws_size,
                              hipStream_t stream) {
    const float* Q  = (const float*)d_in[0];
    const float* K  = (const float*)d_in[1];
    const float* V  = (const float*)d_in[2];
    const float* Wq = (const float*)d_in[3];
    const float* bq = (const float*)d_in[4];
    const float* Wk = (const float*)d_in[5];
    const float* bk = (const float*)d_in[6];
    const float* Wv = (const float*)d_in[7];
    const float* bv = (const float*)d_in[8];
    const float* Wo = (const float*)d_in[9];
    const float* bo = (const float*)d_in[10];
    float* out = (float*)d_out;

    char*  ws  = (char*)d_ws;
    size_t off = 0;
    auto take = [&](size_t elems) {
        __bf16* p = (__bf16*)(ws + off);
        off += elems * sizeof(__bf16);
        off = (off + 255) & ~(size_t)255;
        return p;
    };

    const size_t ND = (size_t)BQ * LQ * DM;
    const size_t WW = (size_t)DM * DM;

    __bf16* Xq  = take(ND);  __bf16* Xk  = take(ND);  __bf16* Xv  = take(ND);
    __bf16* Wqb = take(WW);  __bf16* Wkb = take(WW);
    __bf16* Wvb = take(WW);  __bf16* Wob = take(WW);
    __bf16* qhp = take(ND);  __bf16* khp = take(ND);  __bf16* vTp = take(ND);
    __bf16* attn = take(ND);
    (void)ws_size; (void)in_sizes; (void)n_in; (void)out_size;

    // 1. fp32 -> bf16 conversions
    cvt_f32_bf16<<<1024, 256, 0, stream>>>(Q,  Xq,  (int)ND);
    cvt_f32_bf16<<<1024, 256, 0, stream>>>(K,  Xk,  (int)ND);
    cvt_f32_bf16<<<1024, 256, 0, stream>>>(V,  Xv,  (int)ND);
    cvt_f32_bf16<<<512,  256, 0, stream>>>(Wq, Wqb, (int)WW);
    cvt_f32_bf16<<<512,  256, 0, stream>>>(Wk, Wkb, (int)WW);
    cvt_f32_bf16<<<512,  256, 0, stream>>>(Wv, Wvb, (int)WW);
    cvt_f32_bf16<<<512,  256, 0, stream>>>(Wo, Wob, (int)WW);

    // 2. projections (head-split stores; V stored transposed for P@V)
    dim3 gg(DM / 128, (BQ * LQ) / 128);   // (8, 32)
    gemm_nt_bf16<<<gg, 256, 0, stream>>>(Xq, Wqb, bq, qhp, nullptr,
                                         BQ * LQ, DM, DM, 0);
    gemm_nt_bf16<<<gg, 256, 0, stream>>>(Xk, Wkb, bk, khp, nullptr,
                                         BQ * LQ, DM, DM, 0);
    gemm_nt_bf16<<<gg, 256, 0, stream>>>(Xv, Wvb, bv, vTp, nullptr,
                                         BQ * LQ, DM, DM, 1);

    // 3. flash attention (TDM-staged K/V tiles)
    dim3 ag(LQ / 128, BQ * NH);           // (16, 32)
    flash_attn<<<ag, 128, 0, stream>>>(qhp, khp, vTp, attn);

    // 4. output projection -> fp32 d_out
    gemm_nt_bf16<<<gg, 256, 0, stream>>>(attn, Wob, bo, nullptr, out,
                                         BQ * LQ, DM, DM, 2);
}
